// MedicalVisionTransformer_66692252172331
// MI455X (gfx1250) — matrix operand order repaired and checked by reading, not compiled, not run on hardware
//
#include <hip/hip_runtime.h>
#include <hip/hip_bf16.h>
#include <math.h>

// ---------------- problem constants ----------------
#define LNUM  12
#define HD    768
#define NHN   12
#define DHN   64
#define FFN   3072
#define BN_   32
#define RN    29
#define SN    30
#define DN    14
#define EN    15
#define RKN   8
#define H2N   384
#define MROWS (BN_*SN)   // 960

typedef __attribute__((ext_vector_type(16))) _Float16 v16h;
typedef __attribute__((ext_vector_type(8)))  _Float16 v8h;
typedef __attribute__((ext_vector_type(8)))  float    v8f;
typedef __attribute__((ext_vector_type(4)))  unsigned int v4u;
typedef __attribute__((ext_vector_type(8)))  int v8i;
typedef __attribute__((ext_vector_type(4)))  int v4i;

// ---------------- generic f16-WMMA GEMM: C[M,N] = A[M,K] * W[N,K]^T ----------------
// fp32 operands in memory, converted to f16 while staging into double-buffered LDS.
#define BM 128
#define BN 128
#define BK 32

__global__ __launch_bounds__(256) void gemm_f16(
    const float* __restrict__ A, int lda, long long sA,
    const float* __restrict__ W, int ldw, long long sW,
    float* __restrict__ C, int ldc, long long sC,
    int M, int N, int K)
{
  A += (size_t)blockIdx.z * sA;
  W += (size_t)blockIdx.z * sW;
  C += (size_t)blockIdx.z * sC;

  __shared__ _Float16 As[2][BM * BK];   // [m][k], stride BK
  __shared__ _Float16 Ws[2][BN * BK];   // [n][k], stride BK

  const int tid  = threadIdx.x;
  const int lane = tid & 31;
  const int wave = tid >> 5;
  const int wm   = wave >> 2;        // 0..1  : 64-row half
  const int wn   = wave & 3;         // 0..3  : 32-col quarter
  const int m0   = blockIdx.y * BM;
  const int n0   = blockIdx.x * BN;

  // loader coordinates: both tiles are 128x32, 16 fp32 per thread
  const int lr = tid >> 1;           // 0..127
  const int lc = (tid & 1) * 16;     // 0 or 16

  // branchless bounds: clamp row, zero via mask multiply
  const int   arow = m0 + lr;
  const int   wrow = n0 + lr;
  const float amsk = (arow < M) ? 1.f : 0.f;
  const float wmsk = (wrow < N) ? 1.f : 0.f;
  const float* aptr = A + (size_t)((arow < M) ? arow : (M - 1)) * lda + lc;
  const float* wptr = W + (size_t)((wrow < N) ? wrow : (N - 1)) * ldw + lc;

  v8f acc[4][2] = {};
  float4 ra[4], rw[4];

  const int tiles = K / BK;          // K is always a multiple of 32 here

  // prologue: stage tile 0
  #pragma unroll
  for (int q = 0; q < 4; ++q) {
    ra[q] = *(const float4*)(aptr + q * 4);
    rw[q] = *(const float4*)(wptr + q * 4);
  }
  {
    _Float16* da = &As[0][lr * BK + lc];
    _Float16* dw = &Ws[0][lr * BK + lc];
    #pragma unroll
    for (int q = 0; q < 4; ++q) {
      da[q*4+0] = (_Float16)(ra[q].x * amsk); da[q*4+1] = (_Float16)(ra[q].y * amsk);
      da[q*4+2] = (_Float16)(ra[q].z * amsk); da[q*4+3] = (_Float16)(ra[q].w * amsk);
      dw[q*4+0] = (_Float16)(rw[q].x * wmsk); dw[q*4+1] = (_Float16)(rw[q].y * wmsk);
      dw[q*4+2] = (_Float16)(rw[q].z * wmsk); dw[q*4+3] = (_Float16)(rw[q].w * wmsk);
    }
  }
  __syncthreads();

  for (int kb = 0; kb < tiles; ++kb) {
    const int cur = kb & 1;
    const bool more = (kb + 1 < tiles);

    // issue global loads for next tile first (latency hidden under WMMAs)
    if (more) {
      const float* ap = aptr + (size_t)(kb + 1) * BK;
      const float* wp = wptr + (size_t)(kb + 1) * BK;
      #pragma unroll
      for (int q = 0; q < 4; ++q) {
        ra[q] = *(const float4*)(ap + q * 4);
        rw[q] = *(const float4*)(wp + q * 4);
      }
      if (kb + 2 < tiles) {
        __builtin_prefetch(ap + BK, 0, 1);
        __builtin_prefetch(wp + BK, 0, 1);
      }
    }

    // fragments from current buffer
    v16h afr[4], bfr[2];
    #pragma unroll
    for (int mi = 0; mi < 4; ++mi) {
      // A 16x32 f16 layout: lane<16 holds K 0-7 & 16-23; lane>=16 holds K 8-15 & 24-31
      const _Float16* p = &As[cur][(wm*64 + mi*16 + (lane & 15)) * BK + ((lane < 16) ? 0 : 8)];
      union { v16h v; v8h h[2]; } u;
      u.h[0] = *(const v8h*)p;
      u.h[1] = *(const v8h*)(p + 16);
      afr[mi] = u.v;
    }
    #pragma unroll
    for (int ni = 0; ni < 2; ++ni) {
      // B 32x16 f16 layout: lane holds column n=lane%16, K 0-7 & 16-23 (lane<16) or 8-15 & 24-31
      const _Float16* p = &Ws[cur][(wn*32 + ni*16 + (lane & 15)) * BK + ((lane < 16) ? 0 : 8)];
      union { v16h v; v8h h[2]; } u;
      u.h[0] = *(const v8h*)p;
      u.h[1] = *(const v8h*)(p + 16);
      bfr[ni] = u.v;
    }
    #pragma unroll
    for (int mi = 0; mi < 4; ++mi)
      #pragma unroll
      for (int ni = 0; ni < 2; ++ni)
        acc[mi][ni] = __builtin_amdgcn_wmma_f32_16x16x32_f16(
            false, afr[mi], false, bfr[ni], (short)0, acc[mi][ni], false, false);

    // stage next tile into the other buffer
    if (more) {
      _Float16* da = &As[cur ^ 1][lr * BK + lc];
      _Float16* dw = &Ws[cur ^ 1][lr * BK + lc];
      #pragma unroll
      for (int q = 0; q < 4; ++q) {
        da[q*4+0] = (_Float16)(ra[q].x * amsk); da[q*4+1] = (_Float16)(ra[q].y * amsk);
        da[q*4+2] = (_Float16)(ra[q].z * amsk); da[q*4+3] = (_Float16)(ra[q].w * amsk);
        dw[q*4+0] = (_Float16)(rw[q].x * wmsk); dw[q*4+1] = (_Float16)(rw[q].y * wmsk);
        dw[q*4+2] = (_Float16)(rw[q].z * wmsk); dw[q*4+3] = (_Float16)(rw[q].w * wmsk);
      }
    }
    __syncthreads();
  }

  // store: v8f element r -> row = base + r + (lane>=16 ? 8 : 0), col = lane%16
  #pragma unroll
  for (int mi = 0; mi < 4; ++mi) {
    #pragma unroll
    for (int ni = 0; ni < 2; ++ni) {
      int col   = n0 + wn*32 + ni*16 + (lane & 15);
      int rbase = m0 + wm*64 + mi*16 + ((lane < 16) ? 0 : 8);
      if (col < N) {
        #pragma unroll
        for (int r = 0; r < 8; ++r) {
          int row = rbase + r;
          if (row < M) C[(size_t)row * ldc + col] = acc[mi][ni][r];
        }
      }
    }
  }
}

// ---------------- LayerNorm (affine), one block per row ----------------
__global__ __launch_bounds__(256) void ln_kernel(
    const float* __restrict__ x, const float* __restrict__ w, const float* __restrict__ b,
    float* __restrict__ out, int Hdm, float eps)
{
  int row = blockIdx.x;
  const float* xr = x + (size_t)row * Hdm;
  float* orow = out + (size_t)row * Hdm;
  __shared__ float r1[256], r2[256];
  int tid = threadIdx.x;
  float s1 = 0.f, s2 = 0.f;
  float xv[4];
  int c = 0;
  for (int h = tid; h < Hdm; h += 256) { float v = xr[h]; xv[c++] = v; s1 += v; s2 += v * v; }
  r1[tid] = s1; r2[tid] = s2; __syncthreads();
  for (int off = 128; off > 0; off >>= 1) {
    if (tid < off) { r1[tid] += r1[tid + off]; r2[tid] += r2[tid + off]; }
    __syncthreads();
  }
  float mu  = r1[0] / Hdm;
  float var = r2[0] / Hdm - mu * mu;
  float inv = rsqrtf(var + eps);
  c = 0;
  for (int h = tid; h < Hdm; h += 256) {
    float xn = (xv[c++] - mu) * inv;
    orow[h] = xn * w[h] + b[h];
  }
}

// ---------------- attention core: one block per (batch, head) ----------------
// q/k/v staging is a 3D tile (64 floats x 3 projections x 30 rows) -> TDM to LDS.
__global__ __launch_bounds__(256) void attn_kernel(
    const float* __restrict__ qkv, const float* __restrict__ qb, float* __restrict__ ctx)
{
  int b  = blockIdx.x / NHN;
  int nh = blockIdx.x % NHN;
  __shared__ float sh[SN][3][DHN];     // [s][t][d]
  __shared__ float sc[SN][32];
  int tid = threadIdx.x;

  const float* gbase = qkv + (size_t)(b * SN) * (3 * HD) + nh * DHN;

#if __has_builtin(__builtin_amdgcn_tensor_load_to_lds) && __has_builtin(__builtin_amdgcn_s_wait_tensorcnt)
  if (tid < 32) {   // one wave issues the tensor DMA
    unsigned lds = (unsigned)(uintptr_t)&sh[0][0][0];
    unsigned long long ga = (unsigned long long)(uintptr_t)gbase;
    v4u g0;
    g0.x = 1u;                                              // count=1 (valid user D#)
    g0.y = lds;                                             // lds_addr
    g0.z = (unsigned)(ga & 0xffffffffu);                    // global_addr[31:0]
    g0.w = (unsigned)((ga >> 32) & 0x01ffffffu) | 0x80000000u; // addr[56:32] | type=2
    v8i g1;
    g1[0] = (int)(2u << 16);                                // data_size=4B; mask=0; no pad
    g1[1] = (int)(DHN << 16);                               // tensor_dim0[15:0]=64
    g1[2] = (int)(3 << 16);                                 // dim0 hi=0 | tensor_dim1[15:0]=3
    g1[3] = (int)(DHN << 16);                               // dim1 hi=0 | tile_dim0=64
    g1[4] = (int)(3 | (SN << 16));                          // tile_dim1=3 | tile_dim2=30
    g1[5] = HD;                                             // tensor_dim0_stride lo32 = 768
    g1[6] = (int)((3 * HD) << 16);                          // stride0 hi=0 | stride1 lo16=2304
    g1[7] = (int)((3 * HD) >> 16);                          // stride1[47:16] = 0
    v4i g2;
    g2[0] = SN;                                             // tensor_dim2 = 30
    g2[1] = 1;                                              // tensor_dim3 (unused)
    g2[2] = 0;                                              // dim2_stride lo32
    g2[3] = 0;                                              // dim2_stride hi | tile_dim3=0
    v4i g3 = {0, 0, 0, 0};
    v8i g4 = {0, 0, 0, 0, 0, 0, 0, 0};                      // extra group (clang-23 6-arg form)
    __builtin_amdgcn_tensor_load_to_lds(g0, g1, g2, g3, g4, 0);
    __builtin_amdgcn_s_wait_tensorcnt(0);
  }
  __syncthreads();
#else
  for (int idx = tid; idx < SN * 3 * DHN; idx += 256) {
    int s = idx / (3 * DHN); int rem = idx % (3 * DHN);
    int t = rem / DHN, d = rem % DHN;
    sh[s][t][d] = gbase[(size_t)s * (3 * HD) + t * HD + d];
  }
  __syncthreads();
#endif

  // add qkv bias in place
  for (int idx = tid; idx < SN * 3 * DHN; idx += 256) {
    int s = idx / (3 * DHN); int rem = idx % (3 * DHN);
    int t = rem / DHN, d = rem % DHN;
    sh[s][t][d] += qb[t * HD + nh * DHN + d];
  }
  __syncthreads();

  for (int idx = tid; idx < SN * SN; idx += 256) {
    int qi = idx / SN, ki = idx % SN;
    float s = 0.f;
    #pragma unroll
    for (int d = 0; d < DHN; ++d) s += sh[qi][0][d] * sh[ki][1][d];
    sc[qi][ki] = s * 0.125f;      // 1/sqrt(64)
  }
  __syncthreads();
  for (int qi = tid; qi < SN; qi += 256) {
    float mx = -1e30f;
    for (int ki = 0; ki < SN; ++ki) mx = fmaxf(mx, sc[qi][ki]);
    float ssum = 0.f;
    for (int ki = 0; ki < SN; ++ki) { float e = __expf(sc[qi][ki] - mx); sc[qi][ki] = e; ssum += e; }
    float invs = 1.f / ssum;
    for (int ki = 0; ki < SN; ++ki) sc[qi][ki] *= invs;
  }
  __syncthreads();
  for (int idx = tid; idx < SN * DHN; idx += 256) {
    int qi = idx / DHN, d = idx % DHN;
    float s = 0.f;
    for (int ki = 0; ki < SN; ++ki) s += sc[qi][ki] * sh[ki][2][d];
    ctx[((size_t)(b * SN + qi)) * HD + nh * DHN + d] = s;
  }
}

// ---------------- small elementwise kernels ----------------
__global__ void init_hs(const float* __restrict__ region, const float* __restrict__ ctok,
                        float* __restrict__ hs, int total)
{
  int idx = blockIdx.x * blockDim.x + threadIdx.x;
  if (idx >= total) return;
  int h = idx % HD;
  int s = (idx / HD) % SN;
  int b = idx / (SN * HD);
  hs[idx] = (s == 0) ? ctok[h] : region[((size_t)(b * RN + (s - 1))) * HD + h];
}

__global__ void attn_bias_res(float* __restrict__ ao, const float* __restrict__ ob,
                              const float* __restrict__ hs, float* __restrict__ res1, int total)
{
  int idx = blockIdx.x * blockDim.x + threadIdx.x;
  if (idx >= total) return;
  float v = ao[idx] + ob[idx % HD];
  ao[idx] = v;
  res1[idx] = v + hs[idx];
}

__global__ void gelu_bias(float* __restrict__ x, const float* __restrict__ b, int total, int ncol)
{
  int idx = blockIdx.x * blockDim.x + threadIdx.x;
  if (idx >= total) return;
  float v = x[idx] + b[idx % ncol];
  x[idx] = 0.5f * v * (1.f + erff(v * 0.70710678118654752f));
}

__global__ void bias_res_out(const float* __restrict__ fo, const float* __restrict__ b2,
                             const float* __restrict__ res1, float* __restrict__ hs, int total)
{
  int idx = blockIdx.x * blockDim.x + threadIdx.x;
  if (idx >= total) return;
  hs[idx] = fo[idx] + b2[idx % HD] + res1[idx];
}

__global__ void bias_inplace(float* __restrict__ x, const float* __restrict__ b, int total, int ncol)
{
  int idx = blockIdx.x * blockDim.x + threadIdx.x;
  if (idx >= total) return;
  x[idx] += b[idx % ncol];
}

__global__ void cnt_kernel(const float* __restrict__ mask, float* __restrict__ cnt)
{
  int d = threadIdx.x;
  if (d < DN) {
    float s = 0.f;
    for (int r = 0; r < RN; ++r) s += mask[r * DN + d];
    cnt[d] = s;
  }
}

__global__ __launch_bounds__(256) void pooled_kernel(
    const float* __restrict__ ao, const float* __restrict__ mask,
    const float* __restrict__ cnt, float* __restrict__ pooled)
{
  int b = blockIdx.x / DN, d = blockIdx.x % DN;
  float inv = 1.f / fmaxf(cnt[d], 1.f);
  for (int h = threadIdx.x; h < HD; h += 256) {
    float s = 0.f;
    for (int r = 0; r < RN; ++r) {
      float m = mask[r * DN + d];
      if (m != 0.f) s += ao[((size_t)(b * SN + 1 + r)) * HD + h] * m;
    }
    pooled[((size_t)(b * DN + d)) * HD + h] = s * inv;
  }
}

// M_e[r,r'] = sum_f A_dn[e,r,f] * B_up[e,f,r']   (all layers j at once; je = j*E+e)
__global__ __launch_bounds__(64) void me_kernel(
    const float* __restrict__ Adn, const float* __restrict__ Bup, float* __restrict__ Me)
{
  int je = blockIdx.x;
  int t = threadIdx.x;          // 0..63
  int r = t >> 3, rp = t & 7;
  const float* a  = Adn + ((size_t)je * RKN + r) * FFN;
  const float* bu = Bup + (size_t)je * FFN * RKN + rp;
  float s = 0.f;
  for (int f = 0; f < FFN; ++f) s += a[f] * bu[(size_t)f * RKN];
  Me[(size_t)je * 64 + t] = s;
}

// a_dn += SCALING * M_e @ a_up
__global__ void adn_fix(const float* __restrict__ Mej, const float* __restrict__ aup,
                        float* __restrict__ adn, int total)
{
  int idx = blockIdx.x * blockDim.x + threadIdx.x;
  if (idx >= total) return;
  int m  = idx / (EN * RKN);
  int er = idx % (EN * RKN);
  int e = er / RKN, r = er % RKN;
  const float* me = Mej + e * 64 + r * 8;
  const float* au = aup + (size_t)m * (EN * RKN) + e * RKN;
  float s = 0.f;
  #pragma unroll
  for (int rp = 0; rp < 8; ++rp) s += me[rp] * au[rp];
  adn[idx] += 2.0f * s;
}

// cls head: LN + gelu + dot -> active flag (sigmoid(p)>0.5 <=> p>0, gated by cnt>0)
__global__ __launch_bounds__(128) void cls_head(
    const float* __restrict__ h1, const float* __restrict__ b1,
    const float* __restrict__ lnw, const float* __restrict__ lnb,
    const float* __restrict__ w2, const float* __restrict__ b2,
    const float* __restrict__ cnt, float* __restrict__ active)
{
  int bd = blockIdx.x;
  int d  = bd % DN;
  const float* hrow = h1 + (size_t)bd * H2N;
  __shared__ float r1[128], r2[128];
  int tid = threadIdx.x;
  float xv[3];
  float s1 = 0.f, s2 = 0.f;
  #pragma unroll
  for (int i = 0; i < 3; ++i) {
    int t = tid + i * 128;
    float v = hrow[t] + b1[d * H2N + t];
    xv[i] = v; s1 += v; s2 += v * v;
  }
  r1[tid] = s1; r2[tid] = s2; __syncthreads();
  for (int off = 64; off > 0; off >>= 1) {
    if (tid < off) { r1[tid] += r1[tid + off]; r2[tid] += r2[tid + off]; }
    __syncthreads();
  }
  float mu = r1[0] / H2N, var = r2[0] / H2N - mu * mu, inv = rsqrtf(var + 1e-5f);
  __syncthreads();
  float p = 0.f;
  #pragma unroll
  for (int i = 0; i < 3; ++i) {
    int t = tid + i * 128;
    float xn = (xv[i] - mu) * inv * lnw[d * H2N + t] + lnb[d * H2N + t];
    float g = 0.5f * xn * (1.f + erff(xn * 0.70710678118654752f));
    p += g * w2[d * H2N + t];
  }
  r1[tid] = p; __syncthreads();
  for (int off = 64; off > 0; off >>= 1) {
    if (tid < off) r1[tid] += r1[tid + off];
    __syncthreads();
  }
  if (tid == 0) {
    float preds = r1[0] + b2[d];
    active[bd] = (cnt[d] > 0.f && preds > 0.f) ? 1.f : 0.f;
  }
}

__global__ void gate_kernel(const float* __restrict__ active, const float* __restrict__ mask,
                            float* __restrict__ gate)
{
  int idx = blockIdx.x * blockDim.x + threadIdx.x;
  if (idx >= BN_ * SN) return;
  int b = idx / SN, s = idx % SN;
  float a[EN];
  float sum = 0.f;
  for (int e = 0; e < DN; ++e) {
    float v = 0.f;
    if (s > 0 && mask[(s - 1) * DN + e] > 0.f) v = active[b * DN + e];
    a[e] = v; sum += v;
  }
  a[DN] = 1.f; sum += 1.f;
  float inv = 1.f / fmaxf(sum, 1.f);
  for (int e = 0; e < EN; ++e) gate[(size_t)idx * EN + e] = a[e] * inv;
}

// per-(b,s): hs = sum_e w_e * LN_noaffine(base_out + 2*B_dn[e]@a_dn[e]) + res1
__global__ __launch_bounds__(256) void mix_kernel(
    const float* __restrict__ bout, const float* __restrict__ res1,
    const float* __restrict__ adn, const float* __restrict__ gate,
    const float* __restrict__ Bdn, float* __restrict__ hs)
{
  int m = blockIdx.x;
  int tid = threadIdx.x;
  __shared__ float wg[EN];
  __shared__ float av[EN * RKN];
  __shared__ float r1[256], r2[256];
  if (tid < EN) wg[tid] = gate[(size_t)m * EN + tid];
  if (tid < EN * RKN) av[tid] = adn[(size_t)m * EN * RKN + tid];
  __syncthreads();
  const float* bo = bout + (size_t)m * HD;
  const float* rr = res1 + (size_t)m * HD;
  float accv[3] = {0.f, 0.f, 0.f};
  float base[3];
  #pragma unroll
  for (int i = 0; i < 3; ++i) base[i] = bo[tid + i * 256];
  for (int e = 0; e < EN; ++e) {
    float t[3];
    float s1 = 0.f, s2 = 0.f;
    #pragma unroll
    for (int i = 0; i < 3; ++i) {
      int h = tid + i * 256;
      const float* bd = Bdn + ((size_t)e * HD + h) * RKN;
      const float* ad = &av[e * RKN];
      float lr = 0.f;
      #pragma unroll
      for (int r = 0; r < RKN; ++r) lr += bd[r] * ad[r];
      float v = base[i] + 2.0f * lr;
      t[i] = v; s1 += v; s2 += v * v;
    }
    r1[tid] = s1; r2[tid] = s2; __syncthreads();
    for (int off = 128; off > 0; off >>= 1) {
      if (tid < off) { r1[tid] += r1[tid + off]; r2[tid] += r2[tid + off]; }
      __syncthreads();
    }
    float mu = r1[0] / HD, var = r2[0] / HD - mu * mu, inv = rsqrtf(var + 1e-5f);
    float w_ = wg[e];
    __syncthreads();
    #pragma unroll
    for (int i = 0; i < 3; ++i) accv[i] += w_ * (t[i] - mu) * inv;
  }
  #pragma unroll
  for (int i = 0; i < 3; ++i) {
    int h = tid + i * 256;
    hs[(size_t)m * HD + h] = accv[i] + rr[h];
  }
}

// ---------------- host orchestration ----------------
extern "C" void kernel_launch(void* const* d_in, const int* in_sizes, int n_in,
                              void* d_out, int out_size, void* d_ws, size_t ws_size,
                              hipStream_t stream) {
  (void)in_sizes; (void)n_in; (void)out_size; (void)ws_size;
  const float* region = (const float*)d_in[0];
  const float* maskp  = (const float*)d_in[1];
  const float* ctok   = (const float*)d_in[2];
  const float* ln1w   = (const float*)d_in[3];
  const float* ln1b   = (const float*)d_in[4];
  const float* qkvw   = (const float*)d_in[5];
  const float* qkvb   = (const float*)d_in[6];
  const float* aow    = (const float*)d_in[7];
  const float* aob    = (const float*)d_in[8];
  const float* ln2w   = (const float*)d_in[9];
  const float* ln2b   = (const float*)d_in[10];
  const float* fw1    = (const float*)d_in[11];
  const float* fb1    = (const float*)d_in[12];
  const float* fw2    = (const float*)d_in[13];
  const float* fb2    = (const float*)d_in[14];
  const float* cw1    = (const float*)d_in[15];
  const float* cb1    = (const float*)d_in[16];
  const float* clnw   = (const float*)d_in[17];
  const float* clnb   = (const float*)d_in[18];
  const float* cw2    = (const float*)d_in[19];
  const float* cb2    = (const float*)d_in[20];
  const float* lAu    = (const float*)d_in[21];
  const float* lBu    = (const float*)d_in[22];
  const float* lAd    = (const float*)d_in[23];
  const float* lBd    = (const float*)d_in[24];
  const float* flnw   = (const float*)d_in[25];
  const float* flnb   = (const float*)d_in[26];

  char* wsp = (char*)d_ws;
  auto alloc = [&](size_t nfloats) {
    float* p = (float*)wsp;
    wsp += ((nfloats * sizeof(float) + 255) / 256) * 256;
    return p;
  };
  float* hs     = alloc((size_t)MROWS * HD);
  float* xln    = alloc((size_t)MROWS * HD);
  float* qkv    = alloc((size_t)MROWS * 3 * HD);
  float* ctx    = alloc((size_t)MROWS * HD);
  float* aout   = alloc((size_t)MROWS * HD);
  float* res1   = alloc((size_t)MROWS * HD);
  float* yln    = alloc((size_t)MROWS * HD);
  float* ffh    = alloc((size_t)MROWS * FFN);
  float* ffo    = alloc((size_t)MROWS * HD);
  float* aup    = alloc((size_t)MROWS * EN * RKN);
  float* adn    = alloc((size_t)MROWS * EN * RKN);
  float* pooled = alloc((size_t)BN_ * DN * HD);
  float* h1     = alloc((size_t)BN_ * DN * H2N);
  float* act    = alloc((size_t)BN_ * DN);
  float* gate   = alloc((size_t)MROWS * EN);
  float* Me     = alloc((size_t)(LNUM / 2) * EN * RKN * RKN);
  float* cnt    = alloc((size_t)DN);

  const int totBSH = MROWS * HD;
  const int gBSH   = (totBSH + 255) / 256;
  const int totFF  = MROWS * FFN;
  const int gFF    = (totFF + 255) / 256;

  auto gemm = [&](const float* A, int lda, long long sA,
                  const float* W, int ldw, long long sW,
                  float* C, int ldc, long long sC,
                  int M, int N, int K, int batch) {
    dim3 g((N + BN - 1) / BN, (M + BM - 1) / BM, batch);
    gemm_f16<<<g, 256, 0, stream>>>(A, lda, sA, W, ldw, sW, C, ldc, sC, M, N, K);
  };

  init_hs<<<gBSH, 256, 0, stream>>>(region, ctok, hs, totBSH);
  cnt_kernel<<<1, 64, 0, stream>>>(maskp, cnt);
  me_kernel<<<(LNUM / 2) * EN, 64, 0, stream>>>(lAd, lBu, Me);

  for (int i = 0; i < LNUM; ++i) {
    int j = i / 2;
    bool even = (i % 2 == 0);

    // ln1 -> xln ; QKV GEMM ; attention
    ln_kernel<<<MROWS, 256, 0, stream>>>(hs, ln1w + (size_t)i * HD, ln1b + (size_t)i * HD,
                                         xln, HD, 1e-12f);
    gemm(xln, HD, 0, qkvw + (size_t)i * 3 * HD * HD, HD, 0,
         qkv, 3 * HD, 0, MROWS, 3 * HD, HD, 1);
    attn_kernel<<<BN_ * NHN, 256, 0, stream>>>(qkv, qkvb + (size_t)i * 3 * HD, ctx);

    // out projection + bias + residual
    gemm(ctx, HD, 0, aow + (size_t)i * HD * HD, HD, 0,
         aout, HD, 0, MROWS, HD, HD, 1);
    attn_bias_res<<<gBSH, 256, 0, stream>>>(aout, aob + (size_t)i * HD, hs, res1, totBSH);

    // ln2 -> yln ; FFN
    ln_kernel<<<MROWS, 256, 0, stream>>>(res1, ln2w + (size_t)i * HD, ln2b + (size_t)i * HD,
                                         yln, HD, 1e-12f);
    gemm(yln, HD, 0, fw1 + (size_t)i * FFN * HD, HD, 0,
         ffh, FFN, 0, MROWS, FFN, HD, 1);
    gelu_bias<<<gFF, 256, 0, stream>>>(ffh, fb1 + (size_t)i * FFN, totFF, FFN);
    gemm(ffh, FFN, 0, fw2 + (size_t)i * HD * FFN, FFN, 0,
         ffo, HD, 0, MROWS, HD, FFN, 1);

    if (!even) {
      bias_res_out<<<gBSH, 256, 0, stream>>>(ffo, fb2 + (size_t)i * HD, res1, hs, totBSH);
    } else {
      bias_inplace<<<gBSH, 256, 0, stream>>>(ffo, fb2 + (size_t)i * HD, totBSH, HD);

      // LoRA projections (folded: a_dn gets M_e correction)
      gemm(yln, HD, 0, lAu + (size_t)j * EN * RKN * HD, HD, 0,
           aup, EN * RKN, 0, MROWS, EN * RKN, HD, 1);
      gemm(ffh, FFN, 0, lAd + (size_t)j * EN * RKN * FFN, FFN, 0,
           adn, EN * RKN, 0, MROWS, EN * RKN, FFN, 1);
      {
        int tot = MROWS * EN * RKN;
        adn_fix<<<(tot + 255) / 256, 256, 0, stream>>>(Me + (size_t)j * EN * 64, aup, adn, tot);
      }

      // classification heads -> active flags -> gates
      pooled_kernel<<<BN_ * DN, 256, 0, stream>>>(aout, maskp, cnt, pooled);
      gemm(pooled, DN * HD, HD, cw1 + (size_t)j * DN * H2N * HD, HD, (long long)H2N * HD,
           h1, DN * H2N, H2N, BN_, H2N, HD, DN);
      cls_head<<<BN_ * DN, 128, 0, stream>>>(h1, cb1 + (size_t)j * DN * H2N,
                                             clnw + (size_t)j * DN * H2N,
                                             clnb + (size_t)j * DN * H2N,
                                             cw2 + (size_t)j * DN * H2N,
                                             cb2 + (size_t)j * DN, cnt, act);
      gate_kernel<<<(MROWS + 255) / 256, 256, 0, stream>>>(act, maskp, gate);

      // expert mix: LN + weighted sum + residual
      mix_kernel<<<MROWS, 256, 0, stream>>>(ffo, res1, adn, gate,
                                            lBd + (size_t)j * EN * HD * RKN, hs);
    }
  }

  ln_kernel<<<MROWS, 256, 0, stream>>>(hs, flnw, flnb, (float*)d_out, HD, 1e-12f);
}
